// ChannelizedAxialAttentionHead_82678120448397
// MI455X (gfx1250) — compile-verified, matrix operand order checked by reading, not run on hardware
//
#include <hip/hip_runtime.h>

#define DEVI __device__ __forceinline__

constexpr int NB   = 2;
constexpr int Hc   = 65;
constexpr int Wc   = 65;
constexpr int HP   = Hc + 2;      // padded 67
constexpr int WP   = Wc + 2;      // padded 67
constexpr int CIN  = 1024;
constexpr int Fc   = 512;
constexpr int Gc   = 64;
constexpr int HDc  = 128;
constexpr int HW   = Hc * Wc;     // 4225
constexpr int MR   = NB * HW;     // 8450
constexpr int KENT = 9 * CIN;     // 9216

typedef __attribute__((ext_vector_type(16))) __bf16         v16bf;
typedef __attribute__((ext_vector_type(8)))  float          v8f;
typedef __attribute__((ext_vector_type(8)))  unsigned short ushort8;

union BFrag { v16bf v; ushort8 h[2]; };

DEVI unsigned short f2bf(float f) {
  unsigned int u = __float_as_uint(f);
  u += 0x7FFFu + ((u >> 16) & 1u);
  return (unsigned short)(u >> 16);
}
DEVI float leaky(float x) { return x > 0.f ? x : 0.2f * x; }

// ---------------- pad + convert input to bf16 ----------------
// xp: [NB][HP][WP][CIN] bf16, zero borders
__global__ void xpad_kernel(const float* __restrict__ x, unsigned short* __restrict__ xp) {
  size_t i = (size_t)blockIdx.x * 256 + threadIdx.x;
  const size_t total = (size_t)NB * HP * WP * CIN;
  if (i >= total) return;
  int c = (int)(i % CIN);
  size_t r = i / CIN;
  int wp = (int)(r % WP); r /= WP;
  int hp = (int)(r % HP);
  int n  = (int)(r / HP);
  int h = hp - 1, w = wp - 1;
  unsigned short v = 0;
  if (h >= 0 && h < Hc && w >= 0 && w < Wc)
    v = f2bf(x[((size_t)(n * Hc + h) * Wc + w) * CIN + c]);
  xp[i] = v;
}

// w is [K][Nc] row-major (HWIO flattened); wt is [Nc][K] bf16
__global__ void wcvt_kernel(const float* __restrict__ w, unsigned short* __restrict__ wt, int K, int Nc) {
  size_t i = (size_t)blockIdx.x * 256 + threadIdx.x;
  if (i >= (size_t)K * Nc) return;
  int n = (int)(i / K), k = (int)(i % K);
  wt[i] = f2bf(w[(size_t)k * Nc + n]);
}

// ---------------- entry 3x3 conv, implicit GEMM, WMMA bf16 ----------------
// M=MR, N=Fc, K=9*CIN. Block tile 128x128, 8 waves (4 M x 2 N),
// each wave 32x64 = 8 accumulators. Branch-free inner loop (padded input).
__global__ __launch_bounds__(256)
void entry_conv_kernel(const unsigned short* __restrict__ xp,  // [NB,HP,WP,CIN] bf16
                       const unsigned short* __restrict__ wt,  // [Fc][KENT] bf16
                       const float* __restrict__ bns, const float* __restrict__ bnb,
                       unsigned short* __restrict__ yb)        // [MR][Fc] bf16
{
  const int tid = threadIdx.x;
  const int lane = tid & 31, wid = tid >> 5;
  const int lane16 = lane & 15, half = lane >> 4;
  const int wm = wid & 3, wn = wid >> 2;
  const int rowBase = blockIdx.x * 128 + wm * 32;
  const int colBase = blockIdx.y * 128 + wn * 64;

  int m0 = rowBase + lane16;      int m0c = m0 < MR ? m0 : MR - 1;
  int m1 = rowBase + 16 + lane16; int m1c = m1 < MR ? m1 : MR - 1;
  const int n0 = m0c / HW, r0 = m0c % HW, h0 = r0 / Wc, w0 = r0 % Wc;
  const int n1 = m1c / HW, r1 = m1c % HW, h1 = r1 / Wc, w1 = r1 % Wc;

  v8f acc[2][4] = {};
  const unsigned short* bptr[4];
#pragma unroll
  for (int t = 0; t < 4; ++t)
    bptr[t] = wt + (size_t)(colBase + t * 16 + lane16) * KENT + 8 * half;

  for (int tap = 0; tap < 9; ++tap) {
    const int dy = tap / 3, dx = tap % 3;  // padded coords: +0..2 == -1..+1 with pad 1
    const unsigned short* a0p =
        xp + ((size_t)(n0 * HP + h0 + dy) * WP + (w0 + dx)) * CIN + 8 * half;
    const unsigned short* a1p =
        xp + ((size_t)(n1 * HP + h1 + dy) * WP + (w1 + dx)) * CIN + 8 * half;
    const int kt = tap * CIN;
    for (int c0 = 0; c0 < CIN; c0 += 32) {
      BFrag a0, a1;
      a0.h[0] = *(const ushort8*)(a0p + c0);
      a0.h[1] = *(const ushort8*)(a0p + c0 + 16);
      a1.h[0] = *(const ushort8*)(a1p + c0);
      a1.h[1] = *(const ushort8*)(a1p + c0 + 16);
      const int k = kt + c0;
#pragma unroll
      for (int t = 0; t < 4; ++t) {
        BFrag b;
        b.h[0] = *(const ushort8*)(bptr[t] + k);
        b.h[1] = *(const ushort8*)(bptr[t] + k + 16);
        acc[0][t] = __builtin_amdgcn_wmma_f32_16x16x32_bf16(
            false, a0.v, false, b.v, (short)0, acc[0][t], false, false);
        acc[1][t] = __builtin_amdgcn_wmma_f32_16x16x32_bf16(
            false, a1.v, false, b.v, (short)0, acc[1][t], false, false);
      }
    }
  }
#pragma unroll
  for (int t = 0; t < 4; ++t) {
    int col = colBase + t * 16 + lane16;
    float s = bns[col], bo = bnb[col];
#pragma unroll
    for (int g = 0; g < 2; ++g) {
#pragma unroll
      for (int r = 0; r < 8; ++r) {
        int row = rowBase + g * 16 + r + 8 * half;
        if (row < MR) {
          float v = acc[g][t][r] * s + bo;
          yb[(size_t)row * Fc + col] = f2bf(v > 0.f ? v : 0.f);
        }
      }
    }
  }
}

// ---------------- generic 1x1 conv GEMM, WMMA bf16, BN+ReLU, f32 out ----------------
// grid: (ceil(M/128), Nc/64). Block 128x64, 8 waves (4 M x 2 N), each wave 32x32.
__global__ __launch_bounds__(256)
void gemm_bf16_kernel(const unsigned short* __restrict__ X, int M, int K,
                      const unsigned short* __restrict__ Wt,
                      const float* __restrict__ s, const float* __restrict__ bb,
                      float* __restrict__ out, int ldo, int colOff)
{
  const int tid = threadIdx.x;
  const int lane = tid & 31, wid = tid >> 5;
  const int lane16 = lane & 15, half = lane >> 4;
  const int wm = wid & 3, wn = wid >> 2;
  const int rowBase = blockIdx.x * 128 + wm * 32;
  const int colBase = blockIdx.y * 64 + wn * 32;

  int m0 = rowBase + lane16;      int m0c = m0 < M ? m0 : M - 1;
  int m1 = rowBase + 16 + lane16; int m1c = m1 < M ? m1 : M - 1;
  const unsigned short* a0p = X + (size_t)m0c * K + 8 * half;
  const unsigned short* a1p = X + (size_t)m1c * K + 8 * half;
  const unsigned short* bptr[2];
#pragma unroll
  for (int t = 0; t < 2; ++t)
    bptr[t] = Wt + (size_t)(colBase + t * 16 + lane16) * K + 8 * half;

  v8f acc[2][2] = {};
  for (int k = 0; k < K; k += 32) {
    BFrag a0, a1;
    a0.h[0] = *(const ushort8*)(a0p + k);
    a0.h[1] = *(const ushort8*)(a0p + k + 16);
    a1.h[0] = *(const ushort8*)(a1p + k);
    a1.h[1] = *(const ushort8*)(a1p + k + 16);
#pragma unroll
    for (int t = 0; t < 2; ++t) {
      BFrag b;
      b.h[0] = *(const ushort8*)(bptr[t] + k);
      b.h[1] = *(const ushort8*)(bptr[t] + k + 16);
      acc[0][t] = __builtin_amdgcn_wmma_f32_16x16x32_bf16(
          false, a0.v, false, b.v, (short)0, acc[0][t], false, false);
      acc[1][t] = __builtin_amdgcn_wmma_f32_16x16x32_bf16(
          false, a1.v, false, b.v, (short)0, acc[1][t], false, false);
    }
  }
#pragma unroll
  for (int t = 0; t < 2; ++t) {
    int col = colBase + t * 16 + lane16;
    float sc = s[col], bo = bb[col];
#pragma unroll
    for (int g = 0; g < 2; ++g) {
#pragma unroll
      for (int r = 0; r < 8; ++r) {
        int row = rowBase + g * 16 + r + 8 * half;
        if (row < M) {
          float v = acc[g][t][r] * sc + bo;
          out[(size_t)row * ldo + colOff + col] = v > 0.f ? v : 0.f;
        }
      }
    }
  }
}

// ---------------- fused vertical attention: gram + softmax + apply ----------------
__global__ __launch_bounds__(256)
void vert_attn_kernel(const float* __restrict__ qv, const float* __restrict__ feat,
                      float* __restrict__ sv)
{
  __shared__ float sA[65 * 66];
  __shared__ float sB[65 * 128];
  const int b = blockIdx.x, n = b / Wc, w = b % Wc;
  const int tid = threadIdx.x;
  for (int idx = tid; idx < 65 * 64; idx += 256) {
    int i = idx >> 6, g = idx & 63;
    sB[idx] = qv[((size_t)(n * Hc + i) * Wc + w) * Gc + g];
  }
  __syncthreads();
  for (int idx = tid; idx < 65 * 65; idx += 256) {
    int i = idx / 65, j = idx % 65;
    float a = 0.f;
    for (int g = 0; g < 64; ++g) a += sB[i * 64 + g] * sB[j * 64 + g];
    sA[i * 66 + j] = a;
  }
  __syncthreads();
  if (tid < 65) {
    float mx = -1e30f;
    for (int j = 0; j < 65; ++j) mx = fmaxf(mx, sA[tid * 66 + j]);
    float sum = 0.f;
    for (int j = 0; j < 65; ++j) { float e = __expf(sA[tid * 66 + j] - mx); sA[tid * 66 + j] = e; sum += e; }
    float inv = 1.f / sum;
    for (int j = 0; j < 65; ++j) sA[tid * 66 + j] *= inv;
  }
  __syncthreads();
  for (int c0 = 0; c0 < Fc; c0 += 128) {
    for (int idx = tid; idx < 65 * 128; idx += 256) {
      int j = idx >> 7, cc = idx & 127;
      sB[idx] = feat[((size_t)(n * Hc + j) * Wc + w) * Fc + c0 + cc];
    }
    __syncthreads();
    for (int idx = tid; idx < 65 * 128; idx += 256) {
      int i = idx >> 7, cc = idx & 127;
      float a = 0.f;
      for (int j = 0; j < 65; ++j) a += sA[i * 66 + j] * sB[j * 128 + cc];
      sv[((size_t)(n * Hc + i) * Wc + w) * Fc + c0 + cc] = a;
    }
    __syncthreads();
  }
}

// ---------------- fused horizontal attention (gate_v folded into X) ----------------
__global__ __launch_bounds__(256)
void horiz_attn_kernel(const float* __restrict__ qh, const float* __restrict__ sv,
                       const float* __restrict__ gatev, float* __restrict__ sh)
{
  __shared__ float sA[65 * 66];
  __shared__ float sB[65 * 128];
  __shared__ float sG[Fc];
  const int b = blockIdx.x, n = b / Hc, h = b % Hc;
  const int tid = threadIdx.x;
  const float* qbase = qh + (size_t)(n * Hc + h) * Wc * Gc;
  for (int idx = tid; idx < 65 * 64; idx += 256) sB[idx] = qbase[idx];
  for (int c = tid; c < Fc; c += 256) sG[c] = gatev[(size_t)(n * Hc + h) * Fc + c];
  __syncthreads();
  for (int idx = tid; idx < 65 * 65; idx += 256) {
    int i = idx / 65, j = idx % 65;
    float a = 0.f;
    for (int g = 0; g < 64; ++g) a += sB[i * 64 + g] * sB[j * 64 + g];
    sA[i * 66 + j] = a;
  }
  __syncthreads();
  if (tid < 65) {
    float mx = -1e30f;
    for (int j = 0; j < 65; ++j) mx = fmaxf(mx, sA[tid * 66 + j]);
    float sum = 0.f;
    for (int j = 0; j < 65; ++j) { float e = __expf(sA[tid * 66 + j] - mx); sA[tid * 66 + j] = e; sum += e; }
    float inv = 1.f / sum;
    for (int j = 0; j < 65; ++j) sA[tid * 66 + j] *= inv;
  }
  __syncthreads();
  for (int c0 = 0; c0 < Fc; c0 += 128) {
    for (int idx = tid; idx < 65 * 128; idx += 256) {
      int j = idx >> 7, cc = idx & 127;
      sB[idx] = sv[((size_t)(n * Hc + h) * Wc + j) * Fc + c0 + cc] * sG[c0 + cc];
    }
    __syncthreads();
    for (int idx = tid; idx < 65 * 128; idx += 256) {
      int i = idx >> 7, cc = idx & 127;
      float a = 0.f;
      for (int j = 0; j < 65; ++j) a += sA[i * 66 + j] * sB[j * 128 + cc];
      sh[((size_t)(n * Hc + h) * Wc + i) * Fc + c0 + cc] = a;
    }
    __syncthreads();
  }
}

// ---------------- reductions for channel-attention inputs ----------------
__global__ void rowsum_kernel(const float* __restrict__ src, float* __restrict__ out, int mode) {
  int idx = blockIdx.x * 256 + threadIdx.x;
  if (idx >= NB * 65 * Fc) return;
  int c = idx % Fc, a = (idx / Fc) % 65, n = idx / (Fc * 65);
  float acc = 0.f;
  if (mode == 0) {
    const float* p = src + ((size_t)(n * Hc + a) * Wc) * Fc + c;
    for (int w = 0; w < Wc; ++w) acc += p[(size_t)w * Fc];
  } else {
    const float* p = src + ((size_t)(n * Hc) * Wc + a) * Fc + c;
    for (int h = 0; h < Hc; ++h) acc += p[(size_t)h * Wc * Fc];
  }
  out[idx] = acc * (1.0f / (float)HW);
}

// ---------------- channel-attention MLP ----------------
__global__ __launch_bounds__(128)
void gate_mlp_kernel(const float* __restrict__ in, const float* __restrict__ w0,
                     const float* __restrict__ wh, const float* __restrict__ we,
                     float* __restrict__ gate)
{
  __shared__ float sin_[Fc];
  __shared__ float h0[HDc], h1[HDc];
  const int row = blockIdx.x, t = threadIdx.x;
  for (int c = t; c < Fc; c += HDc) sin_[c] = in[(size_t)row * Fc + c];
  __syncthreads();
  float a = 0.f;
  for (int c = 0; c < Fc; ++c) a += sin_[c] * w0[(size_t)c * HDc + t];
  h0[t] = leaky(a);
  __syncthreads();
  float* cur = h0; float* nxt = h1;
  for (int l = 0; l < 4; ++l) {
    float b = 0.f;
    for (int k = 0; k < HDc; ++k) b += cur[k] * wh[((size_t)l * HDc + k) * HDc + t];
    __syncthreads();
    nxt[t] = leaky(b);
    __syncthreads();
    float* tmp = cur; cur = nxt; nxt = tmp;
  }
  for (int c = t; c < Fc; c += HDc) {
    float b = 0.f;
    for (int k = 0; k < HDc; ++k) b += cur[k] * we[(size_t)k * Fc + c];
    gate[(size_t)row * Fc + c] = 1.f / (1.f + __expf(-b));
  }
}

// x_h = s_h * gate_h (broadcast over h), converted to bf16 for the end GEMM
__global__ void xh_kernel(const float* __restrict__ sh, const float* __restrict__ gateh,
                          unsigned short* __restrict__ xhb) {
  size_t i = (size_t)blockIdx.x * 256 + threadIdx.x;
  if (i >= (size_t)MR * Fc) return;
  int c = (int)(i % Fc);
  size_t pix = i / Fc;
  int w = (int)(pix % Wc), n = (int)(pix / HW);
  xhb[i] = f2bf(sh[i] * gateh[((size_t)n * Wc + w) * Fc + c]);
}

// ---------------- image-level block on ORIGINAL inputs ----------------
__global__ void pool_kernel(const float* __restrict__ x, float* __restrict__ pooled) {
  int i = blockIdx.x * 256 + threadIdx.x;
  if (i >= NB * CIN) return;
  int c = i % CIN, n = i / CIN;
  const float* p = x + (size_t)n * HW * CIN + c;
  float acc = 0.f;
  for (int k = 0; k < HW; ++k) acc += p[(size_t)k * CIN];
  pooled[i] = acc * (1.0f / (float)HW);
}

__global__ void il_kernel(const float* __restrict__ pooled, const float* __restrict__ w,
                          const float* __restrict__ s, const float* __restrict__ b,
                          float* __restrict__ il) {
  int i = blockIdx.x * 256 + threadIdx.x;
  if (i >= NB * Fc) return;
  int f = i % Fc, n = i / Fc;
  const float* p = pooled + (size_t)n * CIN;
  float acc = 0.f;
  for (int c = 0; c < CIN; ++c) acc += p[c] * w[(size_t)c * Fc + f];
  float v = acc * s[f] + b[f];
  il[i] = v > 0.f ? v : 0.f;
}

__global__ void bcast_kernel(const float* __restrict__ il, float* __restrict__ out) {
  size_t i = (size_t)blockIdx.x * 256 + threadIdx.x;
  if (i >= (size_t)MR * Fc) return;
  int f = (int)(i % Fc);
  size_t pix = i / Fc;
  int n = (int)(pix / HW);
  out[pix * (2 * Fc) + Fc + f] = il[(size_t)n * Fc + f];
}

// ---------------- host orchestration ----------------
extern "C" void kernel_launch(void* const* d_in, const int* in_sizes, int n_in,
                              void* d_out, int out_size, void* d_ws, size_t ws_size,
                              hipStream_t stream) {
  (void)in_sizes; (void)n_in; (void)out_size; (void)ws_size;
  const float* x       = (const float*)d_in[0];
  const float* entry_w = (const float*)d_in[1];
  const float* entry_s = (const float*)d_in[2];
  const float* entry_b = (const float*)d_in[3];
  const float* vqk_w   = (const float*)d_in[4];
  const float* vqk_s   = (const float*)d_in[5];
  const float* vqk_b   = (const float*)d_in[6];
  const float* hqk_w   = (const float*)d_in[7];
  const float* hqk_s   = (const float*)d_in[8];
  const float* hqk_b   = (const float*)d_in[9];
  const float* val_w   = (const float*)d_in[10];
  const float* val_s   = (const float*)d_in[11];
  const float* val_b   = (const float*)d_in[12];
  const float* vca_w0  = (const float*)d_in[13];
  const float* vca_wh  = (const float*)d_in[14];
  const float* vca_we  = (const float*)d_in[15];
  const float* hca_w0  = (const float*)d_in[16];
  const float* hca_wh  = (const float*)d_in[17];
  const float* hca_we  = (const float*)d_in[18];
  const float* end_w   = (const float*)d_in[19];
  const float* end_s   = (const float*)d_in[20];
  const float* end_b   = (const float*)d_in[21];
  const float* il_w    = (const float*)d_in[22];
  const float* il_s    = (const float*)d_in[23];
  const float* il_b    = (const float*)d_in[24];
  float* out = (float*)d_out;
  char* ws = (char*)d_ws;

  size_t off = 0;
  auto alloc = [&](size_t bytes) { size_t o = off; off = (off + bytes + 255) & ~(size_t)255; return o; };
  unsigned short* xp      = (unsigned short*)(ws + alloc((size_t)NB * HP * WP * CIN * 2));
  unsigned short* wt_ent  = (unsigned short*)(ws + alloc((size_t)Fc * KENT * 2));
  unsigned short* wt_vqk  = (unsigned short*)(ws + alloc((size_t)Gc * Fc * 2));
  unsigned short* wt_hqk  = (unsigned short*)(ws + alloc((size_t)Gc * Fc * 2));
  unsigned short* wt_val  = (unsigned short*)(ws + alloc((size_t)Fc * Fc * 2));
  unsigned short* wt_end  = (unsigned short*)(ws + alloc((size_t)Fc * Fc * 2));
  unsigned short* yb      = (unsigned short*)(ws + alloc((size_t)MR * Fc * 2));
  float* qv    = (float*)(ws + alloc((size_t)MR * Gc * 4));
  float* qh    = (float*)(ws + alloc((size_t)MR * Gc * 4));
  float* feat  = (float*)(ws + alloc((size_t)MR * Fc * 4));
  float* sv    = (float*)(ws + alloc((size_t)MR * Fc * 4));
  float* sh    = (float*)(ws + alloc((size_t)MR * Fc * 4));
  float* rs    = (float*)(ws + alloc((size_t)NB * 65 * Fc * 4));
  float* gatev = (float*)(ws + alloc((size_t)NB * 65 * Fc * 4));
  float* gateh = (float*)(ws + alloc((size_t)NB * 65 * Fc * 4));
  unsigned short* xhb = (unsigned short*)(ws + alloc((size_t)MR * Fc * 2));
  float* pooled = (float*)(ws + alloc((size_t)NB * CIN * 4));
  float* ilbuf  = (float*)(ws + alloc((size_t)NB * Fc * 4));

  const size_t npad = (size_t)NB * HP * WP * CIN;
  xpad_kernel<<<(unsigned)((npad + 255) / 256), 256, 0, stream>>>(x, xp);
  wcvt_kernel<<<(unsigned)(((size_t)KENT * Fc + 255) / 256), 256, 0, stream>>>(entry_w, wt_ent, KENT, Fc);
  wcvt_kernel<<<(unsigned)(((size_t)Fc * Gc + 255) / 256), 256, 0, stream>>>(vqk_w, wt_vqk, Fc, Gc);
  wcvt_kernel<<<(unsigned)(((size_t)Fc * Gc + 255) / 256), 256, 0, stream>>>(hqk_w, wt_hqk, Fc, Gc);
  wcvt_kernel<<<(unsigned)(((size_t)Fc * Fc + 255) / 256), 256, 0, stream>>>(val_w, wt_val, Fc, Fc);
  wcvt_kernel<<<(unsigned)(((size_t)Fc * Fc + 255) / 256), 256, 0, stream>>>(end_w, wt_end, Fc, Fc);

  const int gridM = (MR + 127) / 128;  // 67
  entry_conv_kernel<<<dim3(gridM, Fc / 128), 256, 0, stream>>>(xp, wt_ent, entry_s, entry_b, yb);

  gemm_bf16_kernel<<<dim3(gridM, Gc / 64), 256, 0, stream>>>(yb, MR, Fc, wt_vqk, vqk_s, vqk_b, qv, Gc, 0);
  gemm_bf16_kernel<<<dim3(gridM, Gc / 64), 256, 0, stream>>>(yb, MR, Fc, wt_hqk, hqk_s, hqk_b, qh, Gc, 0);
  gemm_bf16_kernel<<<dim3(gridM, Fc / 64), 256, 0, stream>>>(yb, MR, Fc, wt_val, val_s, val_b, feat, Fc, 0);

  vert_attn_kernel<<<NB * Wc, 256, 0, stream>>>(qv, feat, sv);
  rowsum_kernel<<<(NB * 65 * Fc + 255) / 256, 256, 0, stream>>>(sv, rs, 0);
  gate_mlp_kernel<<<NB * Hc, 128, 0, stream>>>(rs, vca_w0, vca_wh, vca_we, gatev);

  horiz_attn_kernel<<<NB * Hc, 256, 0, stream>>>(qh, sv, gatev, sh);
  rowsum_kernel<<<(NB * 65 * Fc + 255) / 256, 256, 0, stream>>>(sh, rs, 1);
  gate_mlp_kernel<<<NB * Wc, 128, 0, stream>>>(rs, hca_w0, hca_wh, hca_we, gateh);

  const size_t nxh = (size_t)MR * Fc;
  xh_kernel<<<(unsigned)((nxh + 255) / 256), 256, 0, stream>>>(sh, gateh, xhb);
  gemm_bf16_kernel<<<dim3(gridM, Fc / 64), 256, 0, stream>>>(xhb, MR, Fc, wt_end, end_s, end_b, out, 2 * Fc, 0);

  pool_kernel<<<(NB * CIN + 255) / 256, 256, 0, stream>>>(x, pooled);
  il_kernel<<<(NB * Fc + 255) / 256, 256, 0, stream>>>(pooled, il_w, il_s, il_b, ilbuf);
  bcast_kernel<<<(unsigned)((nxh + 255) / 256), 256, 0, stream>>>(ilbuf, out);
}
